// AgentAttention_90263032693145
// MI455X (gfx1250) — matrix operand order repaired and checked by reading, not compiled, not run on hardware
//
#include <hip/hip_runtime.h>

#define BATCH 16
#define CDIM  256
#define NH    8
#define HD    32
#define NTOK  4096
#define AG    49
#define AGP   64

typedef __attribute__((ext_vector_type(16))) __bf16 v16bf;
typedef __attribute__((ext_vector_type(8)))  float  v8f;

static __device__ inline v8f wmma_bf16(v16bf a, v16bf b, v8f c) {
  return __builtin_amdgcn_wmma_f32_16x16x32_bf16(false, a, false, b, (short)0, c, false, false);
}
static __device__ inline v8f zero8f() {
  v8f r;
#pragma unroll
  for (int t = 0; t < 8; ++t) r[t] = 0.0f;
  return r;
}
static __device__ inline v16bf zero16bf() {
  v16bf r;
#pragma unroll
  for (int t = 0; t < 16; ++t) r[t] = (__bf16)0.0f;
  return r;
}
// A-fragment (16x32 bf16): lane m holds row m; halves at k = hi*8+t and 16+hi*8+t
static __device__ inline v16bf ldA_bf(const __bf16* line, int hi) {
  v16bf r;
  const __bf16* p0 = line + hi * 8;
  const __bf16* p1 = line + 16 + hi * 8;
#pragma unroll
  for (int t = 0; t < 8; ++t) { r[t] = p0[t]; r[8 + t] = p1[t]; }
  return r;
}
// B-fragment (32x16 bf16): lane n holds column n; halves at k = hi*8+t and 16+hi*8+t
static __device__ inline v16bf ldB_bf(const __bf16* line, int hi) {
  v16bf r;
  const __bf16* p0 = line + hi * 8;
  const __bf16* p1 = line + 16 + hi * 8;
#pragma unroll
  for (int t = 0; t < 8; ++t) { r[t] = p0[t]; r[8 + t] = p1[t]; }
  return r;
}
static __device__ inline float half_reduce_max(float v) {
  v = fmaxf(v, __shfl_xor(v, 1));
  v = fmaxf(v, __shfl_xor(v, 2));
  v = fmaxf(v, __shfl_xor(v, 4));
  v = fmaxf(v, __shfl_xor(v, 8));
  return v;
}
static __device__ inline float half_reduce_sum(float v) {
  v += __shfl_xor(v, 1);
  v += __shfl_xor(v, 2);
  v += __shfl_xor(v, 4);
  v += __shfl_xor(v, 8);
  return v;
}

// ---------------- 0. one-shot weight conversion f32 -> bf16 --------------------
__global__ __launch_bounds__(256) void k_cvtw(const float* __restrict__ Wq,
                                              const float* __restrict__ Wkv,
                                              const float* __restrict__ Wp,
                                              void* __restrict__ wqkv_,
                                              void* __restrict__ wp_) {
  __bf16* wqkv = (__bf16*)wqkv_;
  __bf16* wp = (__bf16*)wp_;
  int idx = blockIdx.x * blockDim.x + threadIdx.x;
  if (idx < 3 * CDIM * CDIM) {
    float v = (idx < CDIM * CDIM) ? Wq[idx] : Wkv[idx - CDIM * CDIM];
    wqkv[idx] = (__bf16)v;
  }
  if (idx < CDIM * CDIM) wp[idx] = (__bf16)Wp[idx];
}

// ---------------- 1. transpose x [b,c,n] f32 -> xt [b,n,c] bf16 (LDS tile) ---
__global__ __launch_bounds__(256) void k_transpose(const float* __restrict__ x,
                                                   void* __restrict__ xt_) {
  __bf16* xt = (__bf16*)xt_;
  __shared__ float tile[32][33];
  int b = blockIdx.z;
  int n0 = blockIdx.x * 32;
  int c0 = blockIdx.y * 32;
  int tx = threadIdx.x, ty = threadIdx.y;  // 32 x 8
  const float* xb = x + (size_t)b * CDIM * NTOK;
#pragma unroll
  for (int j = 0; j < 4; ++j)
    tile[ty + 8 * j][tx] = xb[(size_t)(c0 + ty + 8 * j) * NTOK + n0 + tx];
  __syncthreads();
  __bf16* xtb = xt + (size_t)b * NTOK * CDIM;
#pragma unroll
  for (int j = 0; j < 4; ++j)
    xtb[(size_t)(n0 + ty + 8 * j) * CDIM + c0 + tx] = (__bf16)tile[tx][ty + 8 * j];
}

// ---------------- 2. QKV GEMM (16x64 per wave): q,k [b,n,c]; v transposed [b,c,n]
__global__ __launch_bounds__(256) void k_qkv(const void* __restrict__ xt_,
                                             const void* __restrict__ wqkv_,
                                             void* __restrict__ q_, void* __restrict__ k_,
                                             void* __restrict__ vT_) {
  const __bf16* xt = (const __bf16*)xt_;
  const __bf16* wqkv = (const __bf16*)wqkv_;
  __bf16* q = (__bf16*)q_;
  __bf16* k = (__bf16*)k_;
  __bf16* vT = (__bf16*)vT_;
  int b = blockIdx.z;
  int wave = threadIdx.x >> 5, lane = threadIdx.x & 31;
  int m = lane & 15, hi = lane >> 4;
  int mt = blockIdx.x * 4 + (wave & 3);   // 256 M-tiles of 16 rows
  int ns = blockIdx.y * 2 + (wave >> 2);  // 12 N-supertiles of 64 cols (768 total)
  int i0 = mt * 16, co0 = ns * 64;
  const __bf16* Arow = xt + ((size_t)b * NTOK + i0 + m) * CDIM;
  const __bf16* Wl0 = wqkv + (size_t)(co0 + m) * CDIM;
  const __bf16* Wl1 = wqkv + (size_t)(co0 + 16 + m) * CDIM;
  const __bf16* Wl2 = wqkv + (size_t)(co0 + 32 + m) * CDIM;
  const __bf16* Wl3 = wqkv + (size_t)(co0 + 48 + m) * CDIM;
  v8f acc[4];
#pragma unroll
  for (int t = 0; t < 4; ++t) acc[t] = zero8f();
  for (int kk = 0; kk < CDIM; kk += 32) {
    __builtin_prefetch(Arow + kk + 64, 0, 3);
    // all loads first so the scheduler can clause them and issue WMMAs back-to-back
    v16bf a = ldA_bf(Arow + kk, hi);
    v16bf w0 = ldB_bf(Wl0 + kk, hi);
    v16bf w1 = ldB_bf(Wl1 + kk, hi);
    v16bf w2 = ldB_bf(Wl2 + kk, hi);
    v16bf w3 = ldB_bf(Wl3 + kk, hi);
    acc[0] = wmma_bf16(a, w0, acc[0]);
    acc[1] = wmma_bf16(a, w1, acc[1]);
    acc[2] = wmma_bf16(a, w2, acc[2]);
    acc[3] = wmma_bf16(a, w3, acc[3]);
  }
#pragma unroll
  for (int t = 0; t < 4; ++t) {
    int co = co0 + t * 16 + m;
#pragma unroll
    for (int r = 0; r < 8; ++r) {
      int row = i0 + r + 8 * hi;
      float val = acc[t][r];
      if (co < CDIM)
        q[((size_t)b * NTOK + row) * CDIM + co] = (__bf16)val;
      else if (co < 2 * CDIM)
        k[((size_t)b * NTOK + row) * CDIM + (co - CDIM)] = (__bf16)val;
      else
        vT[((size_t)b * CDIM + (co - 2 * CDIM)) * NTOK + row] = (__bf16)val;
    }
  }
}

// ---------------- 3. adaptive-avg-pool q -> agent tokens [b,49,c] bf16 ----------
__global__ __launch_bounds__(256) void k_pool(const void* __restrict__ q_,
                                              void* __restrict__ agent_) {
  const __bf16* q = (const __bf16*)q_;
  __bf16* agent = (__bf16*)agent_;
  int idx = blockIdx.x * blockDim.x + threadIdx.x;
  if (idx >= BATCH * AG * CDIM) return;
  int c = idx % CDIM;
  int a = (idx / CDIM) % AG;
  int b = idx / (CDIM * AG);
  int p = a / 7, qq = a % 7;
  int sy = (p * 64) / 7, ey = ((p + 1) * 64 + 6) / 7;
  int sx = (qq * 64) / 7, ex = ((qq + 1) * 64 + 6) / 7;
  float s = 0.0f;
  for (int y = sy; y < ey; ++y)
    for (int x = sx; x < ex; ++x)
      s += (float)q[((size_t)b * NTOK + y * 64 + x) * CDIM + c];
  s /= (float)((ey - sy) * (ex - sx));
  agent[((size_t)b * AG + a) * CDIM + c] = (__bf16)s;
}

// ---------------- 4. position biases (bilinear 7x7 -> 64x64) -------------------
static __device__ inline float bilin7(const float* img, int y, int x) {
  const float sc = 7.0f / 64.0f;
  float fy = (y + 0.5f) * sc - 0.5f;
  float fx = (x + 0.5f) * sc - 0.5f;
  float fy0 = floorf(fy), fx0 = floorf(fx);
  float wy = fy - fy0, wx = fx - fx0;
  int y0 = min(max((int)fy0, 0), 6), y1 = min(max((int)fy0 + 1, 0), 6);
  int x0 = min(max((int)fx0, 0), 6), x1 = min(max((int)fx0 + 1, 0), 6);
  float v00 = img[y0 * 7 + x0], v01 = img[y0 * 7 + x1];
  float v10 = img[y1 * 7 + x0], v11 = img[y1 * 7 + x1];
  return (1.0f - wy) * ((1.0f - wx) * v00 + wx * v01) +
         wy * ((1.0f - wx) * v10 + wx * v11);
}
__global__ __launch_bounds__(256) void k_bias(const float* __restrict__ an,
                                              const float* __restrict__ na,
                                              const float* __restrict__ ahb,
                                              const float* __restrict__ awb,
                                              const float* __restrict__ hab,
                                              const float* __restrict__ wab,
                                              float* __restrict__ pos_bias,
                                              float* __restrict__ agent_bias) {
  int idx = blockIdx.x * blockDim.x + threadIdx.x;
  if (idx >= NH * AG * NTOK) return;
  int i = idx % NTOK;
  int a = (idx / NTOK) % AG;
  int h = idx / (NTOK * AG);
  int y = i >> 6, x = i & 63;
  float p1 = bilin7(an + (size_t)(h * AG + a) * 49, y, x);
  pos_bias[((size_t)h * AG + a) * NTOK + i] =
      p1 + ahb[(size_t)(h * AG + a) * 64 + y] + awb[(size_t)(h * AG + a) * 64 + x];
  float a1 = bilin7(na + (size_t)(h * AG + a) * 49, y, x);
  agent_bias[((size_t)h * NTOK + i) * AG + a] =
      a1 + hab[(size_t)(h * 64 + y) * AG + a] + wab[(size_t)(h * 64 + x) * AG + a];
}

// ---------------- 5. stage 1: agent->kv, flash-style online softmax ------------
__global__ __launch_bounds__(128) void k_stage1(const void* __restrict__ agent_,
                                                const void* __restrict__ k_,
                                                const void* __restrict__ vT_,
                                                const float* __restrict__ pos_bias,
                                                void* __restrict__ agent_vT_) {
  const __bf16* agent = (const __bf16*)agent_;
  const __bf16* kbuf = (const __bf16*)k_;
  const __bf16* vT = (const __bf16*)vT_;
  __bf16* agent_vT = (__bf16*)agent_vT_;
  __shared__ __align__(16) __bf16 pstage[4][16 * 32];
  int bh = blockIdx.x;
  int b = bh >> 3, h = bh & 7;
  int wave = threadIdx.x >> 5, lane = threadIdx.x & 31;
  int m = lane & 15, hi = lane >> 4;
  const float scale = 0.17677669529663687f;  // 32^-0.5
  int arow = 16 * wave + m;
  v16bf afrag;
  if (arow < AG)
    afrag = ldA_bf(agent + ((size_t)b * AG + arow) * CDIM + h * HD, hi);
  else
    afrag = zero16bf();
  float mrun[8], lrun[8];
#pragma unroll
  for (int r = 0; r < 8; ++r) { mrun[r] = -1e30f; lrun[r] = 0.0f; }
  v8f acc0 = zero8f(), acc1 = zero8f();
  const __bf16* kb = kbuf + (size_t)b * NTOK * CDIM + h * HD;
  const __bf16* vb = vT + ((size_t)b * CDIM + h * HD) * NTOK;
  __bf16* ps = pstage[wave];
  for (int c = 0; c < NTOK; c += 32) {
    v16bf bk0 = ldB_bf(kb + (size_t)(c + m) * CDIM, hi);
    v16bf bk1 = ldB_bf(kb + (size_t)(c + 16 + m) * CDIM, hi);
    v16bf bv0 = ldB_bf(vb + (size_t)m * NTOK + c, hi);
    v16bf bv1 = ldB_bf(vb + (size_t)(16 + m) * NTOK + c, hi);
    v8f s0 = wmma_bf16(afrag, bk0, zero8f());
    v8f s1 = wmma_bf16(afrag, bk1, zero8f());
    float p0[8], p1[8];
#pragma unroll
    for (int r = 0; r < 8; ++r) {
      int a = 16 * wave + r + 8 * hi;
      int ac = (a < AG) ? a : (AG - 1);
      const float* pb = pos_bias + ((size_t)h * AG + ac) * NTOK + c;
      float v0 = s0[r] * scale + pb[m];
      float v1 = s1[r] * scale + pb[16 + m];
      float mx = half_reduce_max(fmaxf(v0, v1));
      float mnew = fmaxf(mrun[r], mx);
      float fsc = __expf(mrun[r] - mnew);
      float e0 = __expf(v0 - mnew);
      float e1 = __expf(v1 - mnew);
      float rs = half_reduce_sum(e0 + e1);
      lrun[r] = lrun[r] * fsc + rs;
      mrun[r] = mnew;
      acc0[r] *= fsc;
      acc1[r] *= fsc;
      p0[r] = e0;
      p1[r] = e1;
    }
#pragma unroll
    for (int r = 0; r < 8; ++r) {
      ps[(r + 8 * hi) * 32 + m] = (__bf16)p0[r];
      ps[(r + 8 * hi) * 32 + 16 + m] = (__bf16)p1[r];
    }
    v16bf pA = ldA_bf(ps + m * 32, hi);
    acc0 = wmma_bf16(pA, bv0, acc0);
    acc1 = wmma_bf16(pA, bv1, acc1);
  }
  __bf16* avt = agent_vT + (size_t)bh * (HD * AGP);
#pragma unroll
  for (int r = 0; r < 8; ++r) {
    int a = 16 * wave + r + 8 * hi;
    float inv = (lrun[r] > 0.0f) ? (1.0f / lrun[r]) : 0.0f;
    float o0 = (a < AG) ? acc0[r] * inv : 0.0f;
    float o1 = (a < AG) ? acc1[r] * inv : 0.0f;
    avt[(size_t)m * AGP + a] = (__bf16)o0;
    avt[(size_t)(16 + m) * AGP + a] = (__bf16)o1;
  }
}

// ---------------- 6. stage 2: query->agent attention ---------------------------
__global__ __launch_bounds__(256) void k_stage2(const void* __restrict__ q_,
                                                const void* __restrict__ agent_,
                                                const void* __restrict__ agent_vT_,
                                                const float* __restrict__ agent_bias,
                                                float* __restrict__ attn) {
  const __bf16* qbuf = (const __bf16*)q_;
  const __bf16* agent = (const __bf16*)agent_;
  const __bf16* avt_all = (const __bf16*)agent_vT_;
  __shared__ __align__(16) __bf16 pstage[8][16 * 64];
  int blk = blockIdx.x;
  int strip = blk & 31;
  int bh = blk >> 5;
  int b = bh >> 3, h = bh & 7;
  int wave = threadIdx.x >> 5, lane = threadIdx.x & 31;
  int m = lane & 15, hi = lane >> 4;
  int i0 = strip * 128 + wave * 16;
  const float scale = 0.17677669529663687f;
  v16bf aq = ldA_bf(qbuf + ((size_t)b * NTOK + i0 + m) * CDIM + h * HD, hi);
  v8f s[4];
#pragma unroll
  for (int t = 0; t < 4; ++t) {
    int a = t * 16 + m;
    v16bf bfr;
    if (a < AG)
      bfr = ldB_bf(agent + ((size_t)b * AG + a) * CDIM + h * HD, hi);
    else
      bfr = zero16bf();
    s[t] = wmma_bf16(aq, bfr, zero8f());
  }
  float p[4][8];
#pragma unroll
  for (int r = 0; r < 8; ++r) {
    int i = i0 + r + 8 * hi;
    const float* ab = agent_bias + ((size_t)h * NTOK + i) * AG;
    float v[4];
    float mx = -1e30f;
#pragma unroll
    for (int t = 0; t < 4; ++t) {
      int a = t * 16 + m;
      v[t] = (a < AG) ? (s[t][r] * scale + ab[a]) : -1e30f;
      mx = fmaxf(mx, v[t]);
    }
    mx = half_reduce_max(mx);
    float sum = 0.0f;
#pragma unroll
    for (int t = 0; t < 4; ++t) {
      p[t][r] = __expf(v[t] - mx);
      sum += p[t][r];
    }
    sum = half_reduce_sum(sum);
    float inv = 1.0f / sum;
#pragma unroll
    for (int t = 0; t < 4; ++t) p[t][r] *= inv;
  }
  __bf16* ps = pstage[wave];
#pragma unroll
  for (int r = 0; r < 8; ++r)
#pragma unroll
    for (int t = 0; t < 4; ++t)
      ps[(r + 8 * hi) * 64 + t * 16 + m] = (__bf16)p[t][r];
  v16bf pA0 = ldA_bf(ps + m * 64, hi);
  v16bf pA1 = ldA_bf(ps + m * 64 + 32, hi);
  const __bf16* avt = avt_all + (size_t)bh * (HD * AGP);
  v16bf bv00 = ldB_bf(avt + (size_t)m * AGP, hi);
  v16bf bv01 = ldB_bf(avt + (size_t)m * AGP + 32, hi);
  v16bf bv10 = ldB_bf(avt + (size_t)(16 + m) * AGP, hi);
  v16bf bv11 = ldB_bf(avt + (size_t)(16 + m) * AGP + 32, hi);
  v8f o0 = wmma_bf16(pA0, bv00, zero8f());
  o0 = wmma_bf16(pA1, bv01, o0);
  v8f o1 = wmma_bf16(pA0, bv10, zero8f());
  o1 = wmma_bf16(pA1, bv11, o1);
#pragma unroll
  for (int r = 0; r < 8; ++r) {
    int i = i0 + r + 8 * hi;
    attn[((size_t)b * NTOK + i) * CDIM + h * HD + m] = o0[r];
    attn[((size_t)b * NTOK + i) * CDIM + h * HD + 16 + m] = o1[r];
  }
}

// ------- 7. depthwise 3x3 conv on v, add to attn, emit bf16 A for projection ---
__global__ __launch_bounds__(256) void k_dwconv(const void* __restrict__ vT_,
                                                const float* __restrict__ Wd,
                                                const float* __restrict__ bd,
                                                const float* __restrict__ attn,
                                                void* __restrict__ attnbf_) {
  const __bf16* vT = (const __bf16*)vT_;
  __bf16* attnbf = (__bf16*)attnbf_;
  int idx = blockIdx.x * blockDim.x + threadIdx.x;
  if (idx >= BATCH * CDIM * NTOK) return;
  int i = idx % NTOK;
  int c = (idx / NTOK) % CDIM;
  int b = idx / (NTOK * CDIM);
  int y = i >> 6, x = i & 63;
  const __bf16* vp = vT + ((size_t)b * CDIM + c) * NTOK;
  float s = bd[c];
#pragma unroll
  for (int ky = 0; ky < 3; ++ky) {
    int yy = y + ky - 1;
    if (yy < 0 || yy > 63) continue;
#pragma unroll
    for (int kx = 0; kx < 3; ++kx) {
      int xx = x + kx - 1;
      if (xx < 0 || xx > 63) continue;
      s += Wd[c * 9 + ky * 3 + kx] * (float)vp[yy * 64 + xx];
    }
  }
  size_t o = ((size_t)b * NTOK + i) * CDIM + c;
  attnbf[o] = (__bf16)(attn[o] + s);
}

// ---------------- 8. output projection (16x64 per wave), writes [b,c,n] --------
__global__ __launch_bounds__(256) void k_proj(const void* __restrict__ attnbf_,
                                              const void* __restrict__ wp_,
                                              const float* __restrict__ bp,
                                              float* __restrict__ out) {
  const __bf16* attnbf = (const __bf16*)attnbf_;
  const __bf16* wp = (const __bf16*)wp_;
  int b = blockIdx.z;
  int wave = threadIdx.x >> 5, lane = threadIdx.x & 31;
  int m = lane & 15, hi = lane >> 4;
  int mt = blockIdx.x * 4 + (wave & 3);   // 256 M-tiles
  int ns = blockIdx.y * 2 + (wave >> 2);  // 4 N-supertiles of 64
  int i0 = mt * 16, co0 = ns * 64;
  const __bf16* Arow = attnbf + ((size_t)b * NTOK + i0 + m) * CDIM;
  const __bf16* Wl0 = wp + (size_t)(co0 + m) * CDIM;
  const __bf16* Wl1 = wp + (size_t)(co0 + 16 + m) * CDIM;
  const __bf16* Wl2 = wp + (size_t)(co0 + 32 + m) * CDIM;
  const __bf16* Wl3 = wp + (size_t)(co0 + 48 + m) * CDIM;
  v8f acc[4];
#pragma unroll
  for (int t = 0; t < 4; ++t) acc[t] = zero8f();
  for (int kk = 0; kk < CDIM; kk += 32) {
    __builtin_prefetch(Arow + kk + 64, 0, 3);
    v16bf a = ldA_bf(Arow + kk, hi);
    v16bf w0 = ldB_bf(Wl0 + kk, hi);
    v16bf w1 = ldB_bf(Wl1 + kk, hi);
    v16bf w2 = ldB_bf(Wl2 + kk, hi);
    v16bf w3 = ldB_bf(Wl3 + kk, hi);
    acc[0] = wmma_bf16(a, w0, acc[0]);
    acc[1] = wmma_bf16(a, w1, acc[1]);
    acc[2] = wmma_bf16(a, w2, acc[2]);
    acc[3] = wmma_bf16(a, w3, acc[3]);
  }
#pragma unroll
  for (int t = 0; t < 4; ++t) {
    int co = co0 + t * 16 + m;
    float bias = bp[co];
    float* ob = out + ((size_t)b * CDIM + co) * NTOK;
#pragma unroll
    for (int r = 0; r < 8; ++r) ob[i0 + r + 8 * hi] = acc[t][r] + bias;
  }
}

extern "C" void kernel_launch(void* const* d_in, const int* in_sizes, int n_in,
                              void* d_out, int out_size, void* d_ws, size_t ws_size,
                              hipStream_t stream) {
  (void)in_sizes; (void)n_in; (void)out_size; (void)ws_size;
  const float* x   = (const float*)d_in[0];
  const float* Wq  = (const float*)d_in[1];
  const float* Wkv = (const float*)d_in[2];
  const float* Wp  = (const float*)d_in[3];
  const float* bp  = (const float*)d_in[4];
  const float* Wd  = (const float*)d_in[5];
  const float* bd  = (const float*)d_in[6];
  const float* an  = (const float*)d_in[7];
  const float* na  = (const float*)d_in[8];
  const float* ahb = (const float*)d_in[9];
  const float* awb = (const float*)d_in[10];
  const float* hab = (const float*)d_in[11];
  const float* wab = (const float*)d_in[12];
  float* out = (float*)d_out;

  char* ws = (char*)d_ws;
  size_t off = 0;
  auto take = [&](size_t bytes) -> void* {
    void* p = ws + off;
    off += (bytes + 255) & ~(size_t)255;
    return p;
  };
  void*  xt     = take((size_t)BATCH * NTOK * CDIM * 2);   // bf16
  void*  qb     = take((size_t)BATCH * NTOK * CDIM * 2);   // bf16
  void*  kb     = take((size_t)BATCH * NTOK * CDIM * 2);   // bf16
  void*  vTb    = take((size_t)BATCH * NTOK * CDIM * 2);   // bf16 [b,c,n]
  void*  ag     = take((size_t)BATCH * AG * CDIM * 2);     // bf16
  void*  avT    = take((size_t)BATCH * NH * HD * AGP * 2); // bf16 [bh,d,a]
  void*  wqkvbf = take((size_t)3 * CDIM * CDIM * 2);       // bf16 [768,256]
  void*  wpbf   = take((size_t)CDIM * CDIM * 2);           // bf16 [256,256]
  void*  attnbf = take((size_t)BATCH * NTOK * CDIM * 2);   // bf16
  float* posb = (float*)take((size_t)NH * AG * NTOK * 4);
  float* agb  = (float*)take((size_t)NH * NTOK * AG * 4);
  float* attn = (float*)take((size_t)BATCH * NTOK * CDIM * 4);

  k_cvtw<<<(3 * CDIM * CDIM + 255) / 256, 256, 0, stream>>>(Wq, Wkv, Wp, wqkvbf, wpbf);
  k_transpose<<<dim3(128, 8, BATCH), dim3(32, 8), 0, stream>>>(x, xt);
  k_qkv<<<dim3(64, 6, BATCH), 256, 0, stream>>>(xt, wqkvbf, qb, kb, vTb);
  k_pool<<<(BATCH * AG * CDIM + 255) / 256, 256, 0, stream>>>(qb, ag);
  k_bias<<<(NH * AG * NTOK + 255) / 256, 256, 0, stream>>>(an, na, ahb, awb, hab, wab,
                                                           posb, agb);
  k_stage1<<<BATCH * NH, 128, 0, stream>>>(ag, kb, vTb, posb, avT);
  k_stage2<<<BATCH * NH * 32, 256, 0, stream>>>(qb, ag, avT, agb, attn);
  k_dwconv<<<(BATCH * CDIM * NTOK) / 256, 256, 0, stream>>>(vTb, Wd, bd, attn, attnbf);
  k_proj<<<dim3(64, 2, BATCH), 256, 0, stream>>>(attnbf, wpbf, bp, out);
}
